// MaskedMGN_22471268893326
// MI455X (gfx1250) — matrix operand order repaired and checked by reading, not compiled, not run on hardware
//
#include <hip/hip_runtime.h>

// ---------------------------------------------------------------------------
// MaskedMGN for MI455X (gfx1250): wave32 + V_WMMA_F32_16X16X32_F16 GEMMs,
// fp16 activation/weight/agg storage (bandwidth-bound @ 23.3 TB/s),
// f32 WMMA accumulation, pk_add_f16 scatter atomics.
// ---------------------------------------------------------------------------

#define N_NODES 50000
#define N_EDGES 800000
#define HID     128
#define TILE    128          // rows per block (8 waves x 16-row WMMA tiles)
#define SA      40           // LDS stride (halfs) for 32-wide K chunks (pad 8)
#define ST      136          // LDS stride (halfs) for 128-wide staging (pad 8)
#define NLAYER  4
#define LN_EPS  1e-5f
#define DT_C    0.01f

typedef _Float16 half_t;
typedef __attribute__((ext_vector_type(16))) _Float16 v16h;
typedef __attribute__((ext_vector_type(8)))  _Float16 v8h;
typedef __attribute__((ext_vector_type(2)))  _Float16 v2h;
typedef __attribute__((ext_vector_type(8)))  float    v8f;

// ---------------- WMMA helpers (ISA 7.12.2 layouts, wave32) ----------------

__device__ __forceinline__ v8f wmma_f16(v16h a, v16h b, v8f c) {
  // (neg_a, A, neg_b, B, c_mod, C, reuse_a, reuse_b)
  return __builtin_amdgcn_wmma_f32_16x16x32_f16(false, a, false, b,
                                                (short)0, c, false, false);
}

// A fragment: 16x32 (MxK) tile, row-major in LDS, stride in halfs (mult of 8).
// lanes 0-15: rows 0-15, K = 0..7 & 16..23 ; lanes 16-31: K = 8..15 & 24..31
__device__ __forceinline__ v16h ld_frag_a(const half_t* tile, int stride) {
  const int lane = threadIdx.x & 31;
  const int r    = lane & 15;
  const int kh   = (lane & 16) >> 1;       // 0 or 8
  const half_t* p = tile + r * stride + kh;
  v16h f;
  ((v8h*)&f)[0] = *(const v8h*)(p);        // K = kh .. kh+7
  ((v8h*)&f)[1] = *(const v8h*)(p + 16);   // K = kh+16 .. kh+23
  return f;
}

// B fragment: 32x16 (KxN). Weights stored transposed Wt[n][k] row-major in LDS.
// lanes 0-15: col n, K = 0..7 & 16..23 ; lanes 16-31: col n, K = 8..15 & 24..31
__device__ __forceinline__ v16h ld_frag_b(const half_t* wt, int stride) {
  const int lane = threadIdx.x & 31;
  const int n    = lane & 15;
  const int kb   = (lane & 16) >> 1;       // 0 or 8
  const half_t* p = wt + n * stride + kb;
  v16h f;
  ((v8h*)&f)[0] = *(const v8h*)(p);
  ((v8h*)&f)[1] = *(const v8h*)(p + 16);
  return f;
}

// One 16-row x 128-col WMMA sweep. B fragments are batched in groups of 4 so
// peak live registers (~64 acc + 8 A + 32 B) stay inside the kernel's VGPR
// allocation -- avoids the allocator collapsing all B-frags onto one register
// pair (which serializes ds_load -> s_wait_dscnt 0 -> wmma per tile).
__device__ __forceinline__ void wmma_row(const half_t* sB, v16h a, v8f* acc) {
#pragma unroll
  for (int g = 0; g < 2; ++g) {
    v16h bf[4];
#pragma unroll
    for (int i = 0; i < 4; ++i)
      bf[i] = ld_frag_b(sB + ((g * 4 + i) << 4) * SA, SA);
#pragma unroll
    for (int i = 0; i < 4; ++i)
      acc[g * 4 + i] = wmma_f16(a, bf[i], acc[g * 4 + i]);
  }
}

// Stage one 128(cols) x 32(K) chunk of a pre-transposed f16 weight matrix
// Wt[n][k] (row stride Kst halfs) into sB. 256 threads, 16 halfs each.
__device__ __forceinline__ void stage_wt_chunk(const half_t* Wt, int Kst, int k0,
                                               half_t* sB, int tid) {
  const int n    = tid >> 1;
  const int part = (tid & 1) << 4;         // 0 or 16
  const half_t* sp = Wt + (size_t)n * Kst + k0 + part;
  half_t* dp = sB + n * SA + part;
  *(v8h*)dp       = *(const v8h*)sp;
  *(v8h*)(dp + 8) = *(const v8h*)(sp + 8);
}

// Store 8 accumulator tiles (16 rows x 128 cols per wave) + bias (+relu) as f16.
__device__ __forceinline__ void store_acc_h(half_t* sOut, int stride,
                                            const v8f* acc, const float* sbias,
                                            int wave, int tid, bool do_relu) {
  const int lane = tid & 31;
  const int n    = lane & 15;
  const int mb   = (lane & 16) >> 1;       // 0 or 8
#pragma unroll
  for (int ct = 0; ct < 8; ++ct) {
#pragma unroll
    for (int v = 0; v < 8; ++v) {
      const int col = (ct << 4) + n;
      float val = acc[ct][v] + sbias[col];
      if (do_relu) val = fmaxf(val, 0.f);
      sOut[((wave << 4) + mb + v) * stride + col] = (half_t)val;
    }
  }
}

// K=128 GEMM from f16 LDS input against pre-transposed f16 weights.
__device__ __forceinline__ void gemm128_from_lds(const half_t* sIn, int strideIn,
                                                 const half_t* Wt, half_t* sB,
                                                 v8f* acc, int wave, int tid) {
  for (int kc = 0; kc < 4; ++kc) {
    stage_wt_chunk(Wt, HID, kc * 32, sB, tid);
    __syncthreads();
    const v16h a = ld_frag_a(sIn + (wave << 4) * strideIn + kc * 32, strideIn);
    wmma_row(sB, a, acc);
    __syncthreads();
  }
}

// Per-row LayerNorm stats from f16 LDS tile (row = tid for tid < TILE).
__device__ __forceinline__ void ln_stats(const half_t* sT_, float* sMu,
                                         float* sRs, int tid) {
  if (tid < TILE) {
    const half_t* rp = sT_ + tid * ST;
    float m = 0.f;
    for (int j = 0; j < HID; ++j) m += (float)rp[j];
    m *= (1.f / HID);
    float v = 0.f;
    for (int j = 0; j < HID; ++j) { float d = (float)rp[j] - m; v += d * d; }
    v *= (1.f / HID);
    sMu[tid] = m;
    sRs[tid] = rsqrtf(v + LN_EPS);
  }
}

// Packed f16 atomic add (FLAT_ATOMIC_PK_ADD_F16 when available, CAS fallback).
__device__ __forceinline__ void atomic_add_h2(half_t* p, float a0, float a1) {
#if defined(__HIP_DEVICE_COMPILE__) && \
    __has_builtin(__builtin_amdgcn_flat_atomic_fadd_v2f16)
  v2h v = {(half_t)a0, (half_t)a1};
  (void)__builtin_amdgcn_flat_atomic_fadd_v2f16((v2h*)p, v);
#else
  unsigned* up = (unsigned*)p;
  unsigned old = *up;
  for (;;) {
    v2h c;
    __builtin_memcpy(&c, &old, 4);
    v2h nv = {(half_t)((float)c[0] + a0), (half_t)((float)c[1] + a1)};
    unsigned nw;
    __builtin_memcpy(&nw, &nv, 4);
    unsigned prev = atomicCAS(up, old, nw);
    if (prev == old) break;
    old = prev;
  }
#endif
}

// -------------------- Weight f32 -> f16 transpose prep ---------------------

__global__ __launch_bounds__(256) void transpose_cvt_kernel(
    const float* __restrict__ src,  // (K x 128) row-major f32
    half_t* __restrict__ dst,       // (128 x K) row-major f16
    int K) {
  const int idx = blockIdx.x * 256 + threadIdx.x;
  if (idx < K * HID) {
    const int k = idx >> 7, n = idx & 127;
    dst[(size_t)n * K + k] = (half_t)src[(size_t)k * HID + n];
  }
}

// ---------------------------- Encoder (CIN small) --------------------------

template <int CIN>
__global__ __launch_bounds__(256) void encoder_kernel(
    const float* __restrict__ xin, const float* __restrict__ W1,
    const float* __restrict__ b1, const half_t* __restrict__ W2t,
    const float* __restrict__ b2, const float* __restrict__ g,
    const float* __restrict__ bt, half_t* __restrict__ outBuf, int Nn) {
  __shared__ float  sX[TILE * CIN];
  __shared__ float  sW1[CIN * HID];
  __shared__ half_t sT[TILE * ST];
  __shared__ half_t sB[HID * SA];
  __shared__ float  sb1[HID], sb2[HID], sg[HID], sbt[HID];
  __shared__ float  sMu[TILE], sRs[TILE];

  const int tid = threadIdx.x, wave = tid >> 5;
  const int row0 = blockIdx.x * TILE;

  for (int i = tid; i < TILE * CIN; i += 256) {
    int r = i / CIN, c = i % CIN;
    int rr = row0 + r; rr = rr < Nn ? rr : Nn - 1;
    sX[r * CIN + c] = xin[(size_t)rr * CIN + c];
  }
  for (int i = tid; i < CIN * HID; i += 256) sW1[i] = W1[i];
  if (tid < HID) { sb1[tid] = b1[tid]; sb2[tid] = b2[tid]; sg[tid] = g[tid]; sbt[tid] = bt[tid]; }
  __syncthreads();

  // layer 1 (K = CIN, scalar) -> f16 LDS
  for (int i = tid; i < TILE * HID; i += 256) {
    const int r = i >> 7, c = i & 127;
    float s = sb1[c];
#pragma unroll
    for (int k = 0; k < CIN; ++k) s += sX[r * CIN + k] * sW1[k * HID + c];
    sT[r * ST + c] = (half_t)fmaxf(s, 0.f);
  }
  __syncthreads();

  // layer 2 (K = 128, WMMA)
  v8f acc[8];
#pragma unroll
  for (int c = 0; c < 8; ++c) acc[c] = {};
  gemm128_from_lds(sT, ST, W2t, sB, acc, wave, tid);
  store_acc_h(sT, ST, acc, sb2, wave, tid, false);   // reuse sT for output
  __syncthreads();

  ln_stats(sT, sMu, sRs, tid);
  __syncthreads();

  const int r = tid >> 1, c0 = (tid & 1) << 6;
  const int row = row0 + r;
  if (row < Nn) {
    const float mu = sMu[r], rs = sRs[r];
    half_t* op = outBuf + (size_t)row * HID;
    for (int j = c0; j < c0 + 64; ++j)
      op[j] = (half_t)(((float)sT[r * ST + j] - mu) * rs * sg[j] + sbt[j]);
  }
}

// ------------------------ Edge processor layer (K=384) ---------------------

__global__ __launch_bounds__(256) void edge_layer_kernel(
    const half_t* __restrict__ eIn, half_t* __restrict__ eOut,
    const half_t* __restrict__ hBuf, half_t* __restrict__ agg,
    const int* __restrict__ eidx, const half_t* __restrict__ W1t,
    const float* __restrict__ b1, const half_t* __restrict__ W2t,
    const float* __restrict__ b2, const float* __restrict__ g,
    const float* __restrict__ bt, int Ee) {
  __shared__ half_t sA[TILE * SA];
  __shared__ half_t sB[HID * SA];
  __shared__ half_t sT[TILE * ST];
  __shared__ int    sSrc[TILE], sDst[TILE];
  __shared__ float  sb1[HID], sb2[HID], sg[HID], sbt[HID];
  __shared__ float  sMu[TILE], sRs[TILE];

  const int tid = threadIdx.x, wave = tid >> 5;
  const int row0 = blockIdx.x * TILE;

  if (tid < TILE) {
    const int e = row0 + tid;
    sSrc[tid] = eidx[e];
    sDst[tid] = eidx[Ee + e];
  }
  if (tid < HID) { sb1[tid] = b1[tid]; sb2[tid] = b2[tid]; sg[tid] = g[tid]; sbt[tid] = bt[tid]; }
  __syncthreads();

  // GEMM1: ein = [e | h[src] | h[dst]], K = 384 streamed in 12 chunks of 32.
  v8f acc[8];
#pragma unroll
  for (int c = 0; c < 8; ++c) acc[c] = {};

  const int r    = tid >> 1;
  const int part = (tid & 1) << 4;           // 0 or 16 (two 16B halves)
  for (int kc = 0; kc < 12; ++kc) {
    const half_t* sp;
    if (kc < 4)      sp = eIn  + (size_t)(row0 + r) * HID + kc * 32 + part;
    else if (kc < 8) sp = hBuf + (size_t)sSrc[r] * HID + (kc - 4) * 32 + part;
    else             sp = hBuf + (size_t)sDst[r] * HID + (kc - 8) * 32 + part;
    half_t* dp = sA + r * SA + part;
    *(v8h*)dp       = *(const v8h*)sp;
    *(v8h*)(dp + 8) = *(const v8h*)(sp + 8);
    if (kc < 11) {                            // prefetch next gather chunk
      const int kn = kc + 1;
      const half_t* np;
      if (kn < 4)      np = eIn  + (size_t)(row0 + r) * HID + kn * 32 + part;
      else if (kn < 8) np = hBuf + (size_t)sSrc[r] * HID + (kn - 4) * 32 + part;
      else             np = hBuf + (size_t)sDst[r] * HID + (kn - 8) * 32 + part;
      __builtin_prefetch((const void*)np, 0, 1);
    }
    stage_wt_chunk(W1t, 3 * HID, kc * 32, sB, tid);
    __syncthreads();
    const v16h a = ld_frag_a(sA + (wave << 4) * SA, SA);
    wmma_row(sB, a, acc);
    __syncthreads();
  }
  store_acc_h(sT, ST, acc, sb1, wave, tid, true);    // relu(t) -> f16 LDS
  __syncthreads();

  // GEMM2: K = 128
  v8f acc2[8];
#pragma unroll
  for (int c = 0; c < 8; ++c) acc2[c] = {};
  gemm128_from_lds(sT, ST, W2t, sB, acc2, wave, tid);
  store_acc_h(sT, ST, acc2, sb2, wave, tid, false);  // reuse sT (reads done)
  __syncthreads();

  ln_stats(sT, sMu, sRs, tid);
  __syncthreads();

  // residual e' = e + LN(..), write f16, pk_f16 scatter-add into agg[dst].
  {
    const int rr = tid >> 1, c0 = (tid & 1) << 6;
    const float mu = sMu[rr], rs = sRs[rr];
    const half_t* ep = eIn + (size_t)(row0 + rr) * HID;
    half_t* eo = eOut + (size_t)(row0 + rr) * HID;
    half_t* ag = agg + (size_t)sDst[rr] * HID;
    for (int j = c0; j < c0 + 64; j += 2) {
      const float v0 = ((float)sT[rr * ST + j]     - mu) * rs * sg[j]     + sbt[j];
      const float v1 = ((float)sT[rr * ST + j + 1] - mu) * rs * sg[j + 1] + sbt[j + 1];
      const float e0 = (float)ep[j] + v0;
      const float e1 = (float)ep[j + 1] + v1;
      eo[j]     = (half_t)e0;
      eo[j + 1] = (half_t)e1;
      atomic_add_h2(ag + j, e0, e1);
    }
  }
}

// ------------------------ Node processor layer (K=256) ---------------------

__global__ __launch_bounds__(256) void node_layer_kernel(
    const half_t* __restrict__ hIn, half_t* __restrict__ hOut,
    const half_t* __restrict__ agg, const half_t* __restrict__ W1t,
    const float* __restrict__ b1, const half_t* __restrict__ W2t,
    const float* __restrict__ b2, const float* __restrict__ g,
    const float* __restrict__ bt, int Nn) {
  __shared__ half_t sA[TILE * SA];
  __shared__ half_t sB[HID * SA];
  __shared__ half_t sT[TILE * ST];
  __shared__ float  sb1[HID], sb2[HID], sg[HID], sbt[HID];
  __shared__ float  sMu[TILE], sRs[TILE];

  const int tid = threadIdx.x, wave = tid >> 5;
  const int row0 = blockIdx.x * TILE;
  if (tid < HID) { sb1[tid] = b1[tid]; sb2[tid] = b2[tid]; sg[tid] = g[tid]; sbt[tid] = bt[tid]; }
  __syncthreads();

  v8f acc[8];
#pragma unroll
  for (int c = 0; c < 8; ++c) acc[c] = {};

  const int r    = tid >> 1;
  const int part = (tid & 1) << 4;
  int rr = row0 + r; rr = rr < Nn ? rr : Nn - 1;
  for (int kc = 0; kc < 8; ++kc) {            // nin = [h | agg], K = 256
    const half_t* sp = (kc < 4)
        ? hIn + (size_t)rr * HID + kc * 32 + part
        : agg + (size_t)rr * HID + (kc - 4) * 32 + part;
    half_t* dp = sA + r * SA + part;
    *(v8h*)dp       = *(const v8h*)sp;
    *(v8h*)(dp + 8) = *(const v8h*)(sp + 8);
    stage_wt_chunk(W1t, 2 * HID, kc * 32, sB, tid);
    __syncthreads();
    const v16h a = ld_frag_a(sA + (wave << 4) * SA, SA);
    wmma_row(sB, a, acc);
    __syncthreads();
  }
  store_acc_h(sT, ST, acc, sb1, wave, tid, true);
  __syncthreads();

  v8f acc2[8];
#pragma unroll
  for (int c = 0; c < 8; ++c) acc2[c] = {};
  gemm128_from_lds(sT, ST, W2t, sB, acc2, wave, tid);
  store_acc_h(sT, ST, acc2, sb2, wave, tid, false);
  __syncthreads();

  ln_stats(sT, sMu, sRs, tid);
  __syncthreads();

  {
    const int r2 = tid >> 1, c0 = (tid & 1) << 6;
    const int row = row0 + r2;
    if (row < Nn) {
      const float mu = sMu[r2], rs = sRs[r2];
      const half_t* hp = hIn + (size_t)row * HID;
      half_t* ho = hOut + (size_t)row * HID;
      for (int j = c0; j < c0 + 64; ++j) {
        const float val = ((float)sT[r2 * ST + j] - mu) * rs * sg[j] + sbt[j];
        ho[j] = (half_t)((float)hp[j] + val);
      }
    }
  }
}

// --------------------------------- Decoder ---------------------------------

__global__ __launch_bounds__(256) void decoder_kernel(
    const half_t* __restrict__ hBuf, const float* __restrict__ xin,
    const half_t* __restrict__ W1t, const float* __restrict__ b1,
    const float* __restrict__ W2, const float* __restrict__ b2,
    float* __restrict__ out, int Nn) {
  __shared__ half_t sA[TILE * SA];
  __shared__ half_t sB[HID * SA];
  __shared__ half_t sT[TILE * ST];
  __shared__ float  sb1[HID], sW2[HID * 3], sb2[4];

  const int tid = threadIdx.x, wave = tid >> 5;
  const int row0 = blockIdx.x * TILE;
  if (tid < HID) sb1[tid] = b1[tid];
  for (int i = tid; i < HID * 3; i += 256) sW2[i] = W2[i];
  if (tid < 3) sb2[tid] = b2[tid];
  __syncthreads();

  v8f acc[8];
#pragma unroll
  for (int c = 0; c < 8; ++c) acc[c] = {};
  const int r    = tid >> 1;
  const int part = (tid & 1) << 4;
  int rr = row0 + r; rr = rr < Nn ? rr : Nn - 1;
  for (int kc = 0; kc < 4; ++kc) {
    const half_t* sp = hBuf + (size_t)rr * HID + kc * 32 + part;
    half_t* dp = sA + r * SA + part;
    *(v8h*)dp       = *(const v8h*)sp;
    *(v8h*)(dp + 8) = *(const v8h*)(sp + 8);
    stage_wt_chunk(W1t, HID, kc * 32, sB, tid);
    __syncthreads();
    const v16h a = ld_frag_a(sA + (wave << 4) * SA, SA);
    wmma_row(sB, a, acc);
    __syncthreads();
  }
  store_acc_h(sT, ST, acc, sb1, wave, tid, true);    // relu(h@W1+b1)
  __syncthreads();

  if (tid < TILE) {
    const int row = row0 + tid;
    if (row < Nn) {
      const half_t* tr = sT + tid * ST;
      float o0 = sb2[0], o1 = sb2[1], o2 = sb2[2];
      for (int k = 0; k < HID; ++k) {
        const float t = (float)tr[k];
        o0 += t * sW2[k * 3 + 0];
        o1 += t * sW2[k * 3 + 1];
        o2 += t * sW2[k * 3 + 2];
      }
      const float x0 = xin[(size_t)row * 6 + 0];
      const float z0 = xin[(size_t)row * 6 + 1];
      const float t0 = xin[(size_t)row * 6 + 2];
      const float m = (x0 * x0 + z0 * z0 <= t0 + DT_C) ? 1.f : 0.f;
      out[(size_t)row * 3 + 0] = o0 * m;
      out[(size_t)row * 3 + 1] = o1 * m;
      out[(size_t)row * 3 + 2] = o2 * m;
    }
  }
}

// --------------------------------- Launch ----------------------------------

static inline size_t align256(size_t x) { return (x + 255) & ~(size_t)255; }

extern "C" void kernel_launch(void* const* d_in, const int* in_sizes, int n_in,
                              void* d_out, int out_size, void* d_ws, size_t ws_size,
                              hipStream_t stream) {
  (void)in_sizes; (void)n_in; (void)out_size; (void)ws_size;

  const float* x     = (const float*)d_in[0];
  const int*   eidx  = (const int*)d_in[1];
  const float* eattr = (const float*)d_in[2];
  const float* ne_W1 = (const float*)d_in[3];
  const float* ne_b1 = (const float*)d_in[4];
  const float* ne_W2 = (const float*)d_in[5];
  const float* ne_b2 = (const float*)d_in[6];
  const float* ne_g  = (const float*)d_in[7];
  const float* ne_bt = (const float*)d_in[8];
  const float* ee_W1 = (const float*)d_in[9];
  const float* ee_b1 = (const float*)d_in[10];
  const float* ee_W2 = (const float*)d_in[11];
  const float* ee_b2 = (const float*)d_in[12];
  const float* ee_g  = (const float*)d_in[13];
  const float* ee_bt = (const float*)d_in[14];
  const float* pe_W1 = (const float*)d_in[15];
  const float* pe_b1 = (const float*)d_in[16];
  const float* pe_W2 = (const float*)d_in[17];
  const float* pe_b2 = (const float*)d_in[18];
  const float* pe_g  = (const float*)d_in[19];
  const float* pe_bt = (const float*)d_in[20];
  const float* pn_W1 = (const float*)d_in[21];
  const float* pn_b1 = (const float*)d_in[22];
  const float* pn_W2 = (const float*)d_in[23];
  const float* pn_b2 = (const float*)d_in[24];
  const float* pn_g  = (const float*)d_in[25];
  const float* pn_bt = (const float*)d_in[26];
  const float* de_W1 = (const float*)d_in[27];
  const float* de_b1 = (const float*)d_in[28];
  const float* de_W2 = (const float*)d_in[29];
  const float* de_b2 = (const float*)d_in[30];

  // workspace carve: h (f16), e (f16), agg (f16), transposed f16 weights
  char* ws = (char*)d_ws;
  size_t off = 0;
  auto carve = [&](size_t bytes) {
    char* p = ws + off;
    off += align256(bytes);
    return p;
  };
  half_t* hBuf   = (half_t*)carve((size_t)N_NODES * HID * sizeof(half_t));
  half_t* eBuf   = (half_t*)carve((size_t)N_EDGES * HID * sizeof(half_t));
  half_t* agg    = (half_t*)carve((size_t)N_NODES * HID * sizeof(half_t));
  half_t* wt_ne2 = (half_t*)carve((size_t)HID * HID * sizeof(half_t));
  half_t* wt_ee2 = (half_t*)carve((size_t)HID * HID * sizeof(half_t));
  half_t* wt_pe1 = (half_t*)carve((size_t)NLAYER * HID * 3 * HID * sizeof(half_t));
  half_t* wt_pe2 = (half_t*)carve((size_t)NLAYER * HID * HID * sizeof(half_t));
  half_t* wt_pn1 = (half_t*)carve((size_t)NLAYER * HID * 2 * HID * sizeof(half_t));
  half_t* wt_pn2 = (half_t*)carve((size_t)NLAYER * HID * HID * sizeof(half_t));
  half_t* wt_de1 = (half_t*)carve((size_t)HID * HID * sizeof(half_t));

  auto cvt = [&](const float* src, half_t* dst, int K) {
    transpose_cvt_kernel<<<(K * HID + 255) / 256, 256, 0, stream>>>(src, dst, K);
  };

  // one-shot weight transpose+convert (f32 KxN -> f16 NxK), ~1 MB total
  cvt(ne_W2, wt_ne2, HID);
  cvt(ee_W2, wt_ee2, HID);
  for (int l = 0; l < NLAYER; ++l) {
    cvt(pe_W1 + (size_t)l * 3 * HID * HID, wt_pe1 + (size_t)l * HID * 3 * HID, 3 * HID);
    cvt(pe_W2 + (size_t)l * HID * HID,     wt_pe2 + (size_t)l * HID * HID,     HID);
    cvt(pn_W1 + (size_t)l * 2 * HID * HID, wt_pn1 + (size_t)l * HID * 2 * HID, 2 * HID);
    cvt(pn_W2 + (size_t)l * HID * HID,     wt_pn2 + (size_t)l * HID * HID,     HID);
  }
  cvt(de_W1, wt_de1, HID);

  const int nBlocks = (N_NODES + TILE - 1) / TILE;   // 391
  const int eBlocks = N_EDGES / TILE;                // 6250

  encoder_kernel<6><<<nBlocks, 256, 0, stream>>>(x, ne_W1, ne_b1, wt_ne2, ne_b2,
                                                 ne_g, ne_bt, hBuf, N_NODES);
  encoder_kernel<2><<<eBlocks, 256, 0, stream>>>(eattr, ee_W1, ee_b1, wt_ee2, ee_b2,
                                                 ee_g, ee_bt, eBuf, N_EDGES);

  for (int l = 0; l < NLAYER; ++l) {
    hipMemsetAsync(agg, 0, (size_t)N_NODES * HID * sizeof(half_t), stream);
    edge_layer_kernel<<<eBlocks, 256, 0, stream>>>(
        eBuf, eBuf, hBuf, agg, eidx,
        wt_pe1 + (size_t)l * HID * 3 * HID, pe_b1 + (size_t)l * HID,
        wt_pe2 + (size_t)l * HID * HID,     pe_b2 + (size_t)l * HID,
        pe_g + (size_t)l * HID,             pe_bt + (size_t)l * HID, N_EDGES);
    node_layer_kernel<<<nBlocks, 256, 0, stream>>>(
        hBuf, hBuf, agg,
        wt_pn1 + (size_t)l * HID * 2 * HID, pn_b1 + (size_t)l * HID,
        wt_pn2 + (size_t)l * HID * HID,     pn_b2 + (size_t)l * HID,
        pn_g + (size_t)l * HID,             pn_bt + (size_t)l * HID, N_NODES);
  }

  decoder_kernel<<<nBlocks, 256, 0, stream>>>(hBuf, x, wt_de1, de_b1, de_W2, de_b2,
                                              (float*)d_out, N_NODES);
}
